// DimeNetModel_18975165513861
// MI455X (gfx1250) — matrix operand order repaired and checked by reading, not compiled, not run on hardware
//
#include <hip/hip_runtime.h>

typedef __bf16 bf16;
typedef __attribute__((ext_vector_type(16))) __bf16 v16bf;
typedef __attribute__((ext_vector_type(8)))  float  v8f;

#define F_ACC  1
#define F_BIAS 2
#define F_SILU 4
#define F_ADD  8

__device__ __constant__ float JLZ[16] = {
    3.14159265358979f,  6.28318530717959f,  9.42477796076938f, 12.56637061435917f,
    4.49340945790906f,  7.72525183693771f, 10.90412165942890f, 14.06619391283147f,
    5.76345919689455f,  9.09501133047636f, 12.32294097056658f, 15.51460301088675f,
    6.98793200050052f, 10.41711854737937f, 13.69802315325025f, 16.92362128521432f};

__device__ __forceinline__ float silu_f(float x) { return x / (1.0f + __expf(-x)); }

__device__ __forceinline__ float jl_f(int l, float x) {
    float s = sinf(x), c = cosf(x);
    float x2 = x * x, x3 = x2 * x, x4 = x2 * x2, x5 = x4 * x;
    switch (l) {
    case 0: return s / x;
    case 1: return s / x2 - c / x;
    case 2: return (3.0f / x3 - 1.0f / x) * s - (3.0f / x2) * c;
    case 3: return (15.0f / x4 - 6.0f / x2) * s - (15.0f / x3 - 1.0f / x) * c;
    default: return (105.0f / x5 - 45.0f / x3 + 1.0f / x) * s - (105.0f / x4 - 10.0f / x2) * c;
    }
}

// ---------------- WMMA GEMM: out[M,N] = act(A[M,K] @ Wp + bias) (+addsrc) ----------------
// One wave per 16-row panel. A fragments loaded as 4x float4 per 32-K step and cvt'd
// to bf16; B fragments are pre-packed so each lane reads one contiguous 32B vector.
// FLAGS is compile-time -> branchless epilogue; full tiles skip the row guard.
template <int K, int N, int FLAGS>
__global__ __launch_bounds__(32) void wgemm(const float* __restrict__ A,
                                            const int* __restrict__ rowidx,
                                            const bf16* __restrict__ W,
                                            const float* __restrict__ bias,
                                            const float* __restrict__ addsrc,
                                            float* __restrict__ out, int M) {
    const int lane = threadIdx.x;
    const int half = lane >> 4;
    const int l16  = lane & 15;
    const int m0   = blockIdx.x * 16;
    const bool full = (m0 + 16 <= M);

    const int arow = m0 + l16;
    const bool av  = arow < M;
    long grow = 0;
    if (av) grow = rowidx ? (long)rowidx[arow] : (long)arow;
    const float* ap = A + grow * K;

    constexpr int KF = K / 32;
    constexpr int NT = N / 16;
    v16bf afrag[KF];
#pragma unroll
    for (int kf = 0; kf < KF; ++kf) {
        const int kb = kf * 32 + half * 16;
        float4 c0 = {0.f, 0.f, 0.f, 0.f}, c1 = c0, c2 = c0, c3 = c0;
        if (av) {
            c0 = *(const float4*)(ap + kb);
            c1 = *(const float4*)(ap + kb + 4);
            c2 = *(const float4*)(ap + kb + 8);
            c3 = *(const float4*)(ap + kb + 12);
        }
        afrag[kf][0] = (bf16)c0.x;  afrag[kf][1] = (bf16)c0.y;
        afrag[kf][2] = (bf16)c0.z;  afrag[kf][3] = (bf16)c0.w;
        afrag[kf][4] = (bf16)c1.x;  afrag[kf][5] = (bf16)c1.y;
        afrag[kf][6] = (bf16)c1.z;  afrag[kf][7] = (bf16)c1.w;
        afrag[kf][8] = (bf16)c2.x;  afrag[kf][9] = (bf16)c2.y;
        afrag[kf][10] = (bf16)c2.z; afrag[kf][11] = (bf16)c2.w;
        afrag[kf][12] = (bf16)c3.x; afrag[kf][13] = (bf16)c3.y;
        afrag[kf][14] = (bf16)c3.z; afrag[kf][15] = (bf16)c3.w;
    }

    const v16bf* Wp = (const v16bf*)W;
#pragma unroll
    for (int nt = 0; nt < NT; ++nt) {
        v8f acc = {0.f, 0.f, 0.f, 0.f, 0.f, 0.f, 0.f, 0.f};
#pragma unroll
        for (int kf = 0; kf < KF; ++kf) {
            v16bf b = Wp[(long)(kf * NT + nt) * 32 + lane];
            acc = __builtin_amdgcn_wmma_f32_16x16x32_bf16(false, afrag[kf], false, b,
                                                          (short)0, acc, false, false);
        }
        const int ocol = nt * 16 + l16;
        const float bv = (FLAGS & F_BIAS) ? bias[ocol] : 0.0f;
        if (full) {
#pragma unroll
            for (int r = 0; r < 8; ++r) {
                const int orow = m0 + r + half * 8;
                const long oi = (long)orow * N + ocol;
                float v = acc[r];
                if (FLAGS & F_ACC)  v += out[oi];
                if (FLAGS & F_BIAS) v += bv;
                if (FLAGS & F_SILU) v = silu_f(v);
                if (FLAGS & F_ADD)  v += addsrc[oi];
                out[oi] = v;
            }
        } else {
#pragma unroll
            for (int r = 0; r < 8; ++r) {
                const int orow = m0 + r + half * 8;
                if (orow < M) {
                    const long oi = (long)orow * N + ocol;
                    float v = acc[r];
                    if (FLAGS & F_ACC)  v += out[oi];
                    if (FLAGS & F_BIAS) v += bv;
                    if (FLAGS & F_SILU) v = silu_f(v);
                    if (FLAGS & F_ADD)  v += addsrc[oi];
                    out[oi] = v;
                }
            }
        }
    }
}

// ---------------- weight pack: f32 [nmat,K,N] -> bf16 WMMA B-fragment layout ----------------
// Packed index within a matrix: ((kf*(N/16) + nt)*32 + lane)*16 + e  holds
// W[(kf*32 + (lane>>4)*16 + e)*N + nt*16 + (lane&15)]
__global__ void k_packW(const float* __restrict__ W, bf16* __restrict__ Wp,
                        int K, int N, long total) {
    long i = (long)blockIdx.x * blockDim.x + threadIdx.x;
    if (i >= total) return;
    const long mn = (long)K * N;
    const long mat = i / mn;
    long j = i - mat * mn;
    const int e = (int)(j & 15); j >>= 4;
    const int lane = (int)(j & 31); j >>= 5;
    const int nt = (int)(j % (N / 16));
    const int kf = (int)(j / (N / 16));
    const int k = kf * 32 + (lane >> 4) * 16 + e;
    const int n = nt * 16 + (lane & 15);
    Wp[i] = (bf16)W[mat * mn + (long)k * N + n];
}

// ---------------- small helper kernels ----------------
__global__ void k_small_gemm(const float* __restrict__ A, const float* __restrict__ W,
                             const float* __restrict__ bias, float* __restrict__ out,
                             int M, int K, int N, int flags) {
    long t = (long)blockIdx.x * blockDim.x + threadIdx.x;
    if (t >= (long)M * N) return;
    int m = (int)(t / N), n = (int)(t % N);
    float s = 0.f;
    for (int k = 0; k < K; ++k) s += A[(long)m * K + k] * W[(long)k * N + n];
    if (flags & F_BIAS) s += bias[n];
    if (flags & F_SILU) s = silu_f(s);
    out[t] = s;
}

__global__ void k_mul(float* __restrict__ a, const float* __restrict__ b, long n) {
    long i = (long)blockIdx.x * blockDim.x + threadIdx.x;
    if (i < n) a[i] *= b[i];
}

__global__ void k_add(float* __restrict__ a, const float* __restrict__ b, long n) {
    long i = (long)blockIdx.x * blockDim.x + threadIdx.x;
    if (i < n) a[i] += b[i];
}

__global__ void k_gatherz(const int* __restrict__ z, const int* __restrict__ e,
                          int* __restrict__ o, int n) {
    int i = blockIdx.x * blockDim.x + threadIdx.x;
    if (i < n) o[i] = z[e[i]];
}

__global__ void k_scatter64(const float* __restrict__ t, const int* __restrict__ ei,
                            float* __restrict__ dst, long n) {
    long i = (long)blockIdx.x * blockDim.x + threadIdx.x;
    if (i >= n) return;
    int e = (int)(i >> 6), c = (int)(i & 63);
    atomicAdd(&dst[(long)ei[e] * 64 + c], t[i]);
}

__global__ void k_pdot(const float* __restrict__ t, const float* __restrict__ w,
                       float* __restrict__ P, int n) {
    int i = blockIdx.x * blockDim.x + threadIdx.x;
    if (i >= n) return;
    const float* row = t + (long)i * 128;
    float s = 0.f;
#pragma unroll
    for (int c = 0; c < 128; ++c) s += row[c] * w[c];
    P[i] += s;
}

__global__ void k_greduce(const float* __restrict__ P, float* __restrict__ out, int G, int Aper) {
    int g = blockIdx.x * blockDim.x + threadIdx.x;
    if (g >= G) return;
    float s = 0.f;
    for (int a = 0; a < Aper; ++a) s += P[(long)g * Aper + a];
    out[g] = s;
}

// per-edge geometry: rbf[E,4] and spherical rbf_sph[E,16]
__global__ void k_geom(const float* __restrict__ pos, const int* __restrict__ ei,
                       const int* __restrict__ ej, const float* __restrict__ freq,
                       float* __restrict__ rbf, float* __restrict__ rbfsph, int E) {
    int e = blockIdx.x * blockDim.x + threadIdx.x;
    if (e >= E) return;
    int i = ei[e], j = ej[e];
    float dx = pos[3 * i + 0] - pos[3 * j + 0];
    float dy = pos[3 * i + 1] - pos[3 * j + 1];
    float dz = pos[3 * i + 2] - pos[3 * j + 2];
    float dist = sqrtf(dx * dx + dy * dy + dz * dz);
    float d = dist * 0.25f;  // /CUTOFF
    float env = 0.f;
    if (d < 1.0f) {
        float d2 = d * d, d4 = d2 * d2, d5 = d4 * d, d6 = d5 * d, d7 = d6 * d;
        env = 1.0f / d - 28.0f * d5 + 48.0f * d6 - 21.0f * d7;
    }
#pragma unroll
    for (int n = 0; n < 4; ++n) rbf[(long)e * 4 + n] = env * sinf(freq[n] * d);
    const float SQ2 = 1.41421356237309f;
#pragma unroll
    for (int l = 0; l < 4; ++l) {
#pragma unroll
        for (int n = 0; n < 4; ++n) {
            float zz = JLZ[l * 4 + n];
            float norm = SQ2 / fabsf(jl_f(l + 1, zz));
            rbfsph[(long)e * 16 + l * 4 + n] = env * norm * jl_f(l, zz * d);
        }
    }
}

// per-triplet: angle -> cbf -> sbf -> contraction with S[16,32] -> scatter-add into agg[idx_ji]
__global__ void k_triplet(const float* __restrict__ pos, const int* __restrict__ ii,
                          const int* __restrict__ ij, const int* __restrict__ ik,
                          const int* __restrict__ ikj, const int* __restrict__ iji,
                          const float* __restrict__ rbfsph, const float* __restrict__ S,
                          const float* __restrict__ m32, float* __restrict__ agg, int T) {
    __shared__ float Ssh[512];
    for (int s = threadIdx.x; s < 512; s += blockDim.x) Ssh[s] = S[s];
    __syncthreads();
    int t = blockIdx.x * blockDim.x + threadIdx.x;
    if (t >= T) return;
    int i = ii[t], j = ij[t], k = ik[t];
    float jx = pos[3 * j + 0] - pos[3 * i + 0];
    float jy = pos[3 * j + 1] - pos[3 * i + 1];
    float jz = pos[3 * j + 2] - pos[3 * i + 2];
    float kx = pos[3 * k + 0] - pos[3 * j + 0];
    float ky = pos[3 * k + 1] - pos[3 * j + 1];
    float kz = pos[3 * k + 2] - pos[3 * j + 2];
    float a = jx * kx + jy * ky + jz * kz;
    float cx = jy * kz - jz * ky, cy = jz * kx - jx * kz, cz = jx * ky - jy * kx;
    float b2 = cx * cx + cy * cy + cz * cz;
    float ct = a * rsqrtf(a * a + b2 + 1e-30f);  // cos(atan2(|cross|, dot))
    float cbf[4];
    cbf[0] = 0.28209479177387814f;
    cbf[1] = 0.4886025119029199f * ct;
    cbf[2] = 0.31539156525252005f * (3.0f * ct * ct - 1.0f);
    cbf[3] = 0.3731763325901154f * (5.0f * ct * ct * ct - 3.0f * ct);
    int kj = ikj[t], ji = iji[t];
    float sbf[16];
    const float* rs = rbfsph + (long)kj * 16;
#pragma unroll
    for (int l = 0; l < 4; ++l)
#pragma unroll
        for (int n = 0; n < 4; ++n) sbf[l * 4 + n] = rs[l * 4 + n] * cbf[l];
    const float* mr = m32 + (long)kj * 32;
    float* ar = agg + (long)ji * 32;
#pragma unroll
    for (int c = 0; c < 32; ++c) {
        float f = 0.f;
#pragma unroll
        for (int s = 0; s < 16; ++s) f += sbf[s] * Ssh[s * 32 + c];
        atomicAdd(&ar[c], mr[c] * f);
    }
}

// ---------------- host orchestration ----------------
extern "C" void kernel_launch(void* const* d_in, const int* in_sizes, int n_in,
                              void* d_out, int out_size, void* d_ws, size_t ws_size,
                              hipStream_t stream) {
    (void)n_in; (void)ws_size;
    // input order: top-level insertion order, params flattened jax-style (sorted keys)
    const float* pos       = (const float*)d_in[0];
    const float* emb_atom  = (const float*)d_in[1];
    const float* emb_lin_b = (const float*)d_in[2];
    const float* emb_lin_w = (const float*)d_in[3];
    const float* emb_rbf_b = (const float*)d_in[4];
    const float* emb_rbf_w = (const float*)d_in[5];
    const float* freq      = (const float*)d_in[6];
    const float* as_b  = (const float*)d_in[7];
    const float* as_w  = (const float*)d_in[8];
    const float* bs_b  = (const float*)d_in[9];
    const float* bs_w  = (const float*)d_in[10];
    const float* dwn   = (const float*)d_in[11];
    const float* ji_b  = (const float*)d_in[12];
    const float* ji_w  = (const float*)d_in[13];
    const float* kj_b  = (const float*)d_in[14];
    const float* kj_w  = (const float*)d_in[15];
    const float* lin_b = (const float*)d_in[16];
    const float* lin_w = (const float*)d_in[17];
    const float* rbf1  = (const float*)d_in[18];
    const float* rbf2  = (const float*)d_in[19];
    const float* sbf1  = (const float*)d_in[20];
    const float* sbf2  = (const float*)d_in[21];
    const float* upw   = (const float*)d_in[22];
    const float* o_lin_w  = (const float*)d_in[23];
    const float* o_lins_b = (const float*)d_in[24];
    const float* o_lins_w = (const float*)d_in[25];
    const float* o_rbf_w  = (const float*)d_in[26];
    const float* o_up_w   = (const float*)d_in[27];
    const float* o0_lin_w  = (const float*)d_in[28];
    const float* o0_lins_b = (const float*)d_in[29];
    const float* o0_lins_w = (const float*)d_in[30];
    const float* o0_rbf_w  = (const float*)d_in[31];
    const float* o0_up_w   = (const float*)d_in[32];
    const int* z      = (const int*)d_in[33];
    const int* edge_i = (const int*)d_in[35];
    const int* edge_j = (const int*)d_in[36];
    const int* idx_i  = (const int*)d_in[37];
    const int* idx_j  = (const int*)d_in[38];
    const int* idx_k  = (const int*)d_in[39];
    const int* idx_kj = (const int*)d_in[40];
    const int* idx_ji = (const int*)d_in[41];

    const int Nn = in_sizes[33];
    const int E  = in_sizes[35];
    const int T  = in_sizes[40];
    const int G  = out_size;
    float* outp  = (float*)d_out;

    char* wsp = (char*)d_ws;
    size_t used = 0;
    auto alloc = [&](size_t bytes) -> void* {
        void* p = wsp + used;
        used = (used + bytes + 255) & ~(size_t)255;
        return p;
    };
    auto g1 = [](long n) { return dim3((unsigned)((n + 255) / 256)); };
    // pack weights into WMMA B-fragment layout (bf16), nmat contiguous [K,N] matrices
    auto pack = [&](const float* src, int K, int N, int nmat) -> const bf16* {
        long n = (long)nmat * K * N;
        bf16* p = (bf16*)alloc((size_t)n * sizeof(bf16));
        k_packW<<<g1(n), 256, 0, stream>>>(src, p, K, N, n);
        return p;
    };

    const bf16* b_emb_lin = pack(emb_lin_w, 64, 64, 3);  // 3 contiguous 64x64 row-chunks
    const bf16* b_ji   = pack(ji_w,  64, 64, 4);
    const bf16* b_kj   = pack(kj_w,  64, 64, 4);
    const bf16* b_down = pack(dwn,   64, 32, 4);
    const bf16* b_up   = pack(upw,   32, 64, 4);
    const bf16* b_bs   = pack(bs_w,  64, 64, 8);
    const bf16* b_lin  = pack(lin_w, 64, 64, 4);
    const bf16* b_as   = pack(as_w,  64, 64, 16);
    const bf16* b_oup  = pack(o_up_w,   64, 128, 4);
    const bf16* b_olin = pack(o_lins_w, 128, 128, 12);
    const bf16* b_o0up = pack(o0_up_w,   64, 128, 1);
    const bf16* b_o0ln = pack(o0_lins_w, 128, 128, 3);

    // activation workspace
    float* rbf    = (float*)alloc((size_t)E * 4 * 4);
    float* rbfsph = (float*)alloc((size_t)E * 16 * 4);
    float* fac    = (float*)alloc((size_t)E * 64 * 4);
    float* xb     = (float*)alloc((size_t)E * 64 * 4);
    float* t1     = (float*)alloc((size_t)E * 64 * 4);
    float* t2     = (float*)alloc((size_t)E * 64 * 4);
    float* t3     = (float*)alloc((size_t)E * 64 * 4);
    float* m32    = (float*)alloc((size_t)E * 32 * 4);
    float* agg    = (float*)alloc((size_t)E * 32 * 4);
    float* Rbuf   = (float*)alloc(4 * 64 * 4);
    float* Sbuf   = (float*)alloc(16 * 32 * 4);
    float* tnode  = (float*)alloc((size_t)Nn * 64 * 4);
    float* tup    = (float*)alloc((size_t)Nn * 128 * 4);
    float* ttmp   = (float*)alloc((size_t)Nn * 128 * 4);
    float* P      = (float*)alloc((size_t)Nn * 4);
    int*   zi     = (int*)alloc((size_t)E * 4);
    int*   zj     = (int*)alloc((size_t)E * 4);

    const dim3 gE((unsigned)((E + 15) / 16)), gN((unsigned)((Nn + 15) / 16));
    const long E64 = (long)E * 64;

    // geometry + radial bases
    k_geom<<<g1(E), 256, 0, stream>>>(pos, edge_i, edge_j, freq, rbf, rbfsph, E);

    // h_rbf = silu(rbf @ emb.rbf_w + b)   [E,64] (K=4 -> VALU)
    k_small_gemm<<<g1(E64), 256, 0, stream>>>(rbf, emb_rbf_w, emb_rbf_b, fac, E, 4, 64, F_BIAS | F_SILU);

    // embedding: x = silu([hz_i | hz_j | h_rbf] @ lin_w + b) via 3 accumulating WMMA passes
    k_gatherz<<<g1(E), 256, 0, stream>>>(z, edge_i, zi, E);
    k_gatherz<<<g1(E), 256, 0, stream>>>(z, edge_j, zj, E);
    wgemm<64, 64, 0><<<gE, 32, 0, stream>>>(emb_atom, zi, b_emb_lin, nullptr, nullptr, xb, E);
    wgemm<64, 64, F_ACC><<<gE, 32, 0, stream>>>(emb_atom, zj, b_emb_lin + 64 * 64, nullptr, nullptr, xb, E);
    wgemm<64, 64, F_ACC | F_BIAS | F_SILU><<<gE, 32, 0, stream>>>(fac, nullptr, b_emb_lin + 2 * 64 * 64,
                                                                  emb_lin_b, nullptr, xb, E);

    hipMemsetAsync(P, 0, (size_t)Nn * 4, stream);

    auto out_block = [&](const float* rbfw, const bf16* upw16, const bf16* lins16,
                         const float* linsb, const float* linw) {
        k_small_gemm<<<g1(E64), 256, 0, stream>>>(rbf, rbfw, nullptr, t3, E, 4, 64, 0);
        k_mul<<<g1(E64), 256, 0, stream>>>(t3, xb, E64);
        hipMemsetAsync(tnode, 0, (size_t)Nn * 64 * 4, stream);
        k_scatter64<<<g1(E64), 256, 0, stream>>>(t3, edge_i, tnode, E64);
        wgemm<64, 128, 0><<<gN, 32, 0, stream>>>(tnode, nullptr, upw16, nullptr, nullptr, tup, Nn);
        float* a = tup; float* b = ttmp;
        for (int k = 0; k < 3; ++k) {
            wgemm<128, 128, F_BIAS | F_SILU><<<gN, 32, 0, stream>>>(a, nullptr,
                lins16 + (long)k * 128 * 128, linsb + k * 128, nullptr, b, Nn);
            float* tmp = a; a = b; b = tmp;
        }
        k_pdot<<<g1(Nn), 256, 0, stream>>>(a, linw, P, Nn);
    };

    out_block(o0_rbf_w, b_o0up, b_o0ln, o0_lins_b, o0_lin_w);

    for (int blk = 0; blk < 4; ++blk) {
        // x_ji, x_kj
        wgemm<64, 64, F_BIAS | F_SILU><<<gE, 32, 0, stream>>>(xb, nullptr, b_ji + (long)blk * 4096,
                                                              ji_b + blk * 64, nullptr, t1, E);
        wgemm<64, 64, F_BIAS | F_SILU><<<gE, 32, 0, stream>>>(xb, nullptr, b_kj + (long)blk * 4096,
                                                              kj_b + blk * 64, nullptr, t2, E);
        // R = rbf1@rbf2 [4,64]; fac = rbf@R; x_kj *= fac
        k_small_gemm<<<g1(256), 256, 0, stream>>>(rbf1 + blk * 32, rbf2 + (long)blk * 512, nullptr,
                                                  Rbuf, 4, 8, 64, 0);
        k_small_gemm<<<g1(E64), 256, 0, stream>>>(rbf, Rbuf, nullptr, fac, E, 4, 64, 0);
        k_mul<<<g1(E64), 256, 0, stream>>>(t2, fac, E64);
        // m = silu(x_kj @ down)  [E,32]
        wgemm<64, 32, F_SILU><<<gE, 32, 0, stream>>>(t2, nullptr, b_down + (long)blk * 2048,
                                                     nullptr, nullptr, m32, E);
        // S = sbf1@sbf2 [16,32]; triplet gather/contract/scatter
        k_small_gemm<<<g1(512), 256, 0, stream>>>(sbf1 + blk * 128, sbf2 + (long)blk * 256, nullptr,
                                                  Sbuf, 16, 8, 32, 0);
        hipMemsetAsync(agg, 0, (size_t)E * 32 * 4, stream);
        k_triplet<<<g1(T), 256, 0, stream>>>(pos, idx_i, idx_j, idx_k, idx_kj, idx_ji,
                                             rbfsph, Sbuf, m32, agg, T);
        // x_kj = silu(agg @ up); h = x_ji + x_kj  (in t2)
        wgemm<32, 64, F_SILU><<<gE, 32, 0, stream>>>(agg, nullptr, b_up + (long)blk * 2048,
                                                     nullptr, nullptr, t2, E);
        k_add<<<g1(E64), 256, 0, stream>>>(t2, t1, E64);
        // pre-skip residual: t1 = t2 + silu(silu(t2@bs0+b0)@bs1+b1)
        wgemm<64, 64, F_BIAS | F_SILU><<<gE, 32, 0, stream>>>(t2, nullptr, b_bs + (long)blk * 8192,
                                                              bs_b + blk * 128, nullptr, t3, E);
        wgemm<64, 64, F_BIAS | F_SILU | F_ADD><<<gE, 32, 0, stream>>>(t3, nullptr,
            b_bs + (long)blk * 8192 + 4096, bs_b + blk * 128 + 64, t2, t1, E);
        // skip: t2 = silu(t1@lin+b) + x ; x <- t2
        wgemm<64, 64, F_BIAS | F_SILU | F_ADD><<<gE, 32, 0, stream>>>(t1, nullptr,
            b_lin + (long)blk * 4096, lin_b + blk * 64, xb, t2, E);
        { float* tmp = xb; xb = t2; t2 = tmp; }
        // post-skip residuals
        for (int k = 0; k < 2; ++k) {
            const bf16* w0 = b_as + ((long)blk * 4 + 2 * k) * 4096;
            const float* bb = as_b + ((long)blk * 2 + k) * 128;
            wgemm<64, 64, F_BIAS | F_SILU><<<gE, 32, 0, stream>>>(xb, nullptr, w0, bb, nullptr, t3, E);
            wgemm<64, 64, F_BIAS | F_SILU | F_ADD><<<gE, 32, 0, stream>>>(t3, nullptr, w0 + 4096,
                                                                          bb + 64, xb, t1, E);
            { float* tmp = xb; xb = t1; t1 = tmp; }
        }
        // output block for this layer
        out_block(o_rbf_w + (long)blk * 256, b_oup + (long)blk * 64 * 128,
                  b_olin + (long)blk * 3 * 128 * 128, o_lins_b + (long)blk * 3 * 128,
                  o_lin_w + (long)blk * 128);
    }

    // per-graph reduction (atoms per graph are contiguous in node order)
    k_greduce<<<g1(G), 256, 0, stream>>>(P, outp, G, Nn / G);
}
